// EGNN_51264729645344
// MI455X (gfx1250) — compile-verified, matrix operand order checked by reading, not run on hardware
//
#include <hip/hip_runtime.h>
#include <hip/hip_bf16.h>
#include <stdint.h>

#define N_EDGES 640000
#define NN 40000
#define HID 128
#define IN_CH 64

typedef __bf16 bf16;
typedef __attribute__((ext_vector_type(16))) __bf16 bf16x16;
typedef __attribute__((ext_vector_type(8)))  __bf16 bf16x8;
typedef __attribute__((ext_vector_type(8)))  float   f32x8;

union BF16x16 { bf16x16 v; bf16x8 h[2]; };

__device__ __forceinline__ f32x8 f32x8_zero() {
  f32x8 z = {0.f, 0.f, 0.f, 0.f, 0.f, 0.f, 0.f, 0.f};
  return z;
}

__device__ __forceinline__ f32x8 wmma_bf16(bf16x16 a, bf16x16 b, f32x8 c) {
  return __builtin_amdgcn_wmma_f32_16x16x32_bf16(false, a, false, b, (short)0, c, false, false);
}

__device__ __forceinline__ float silu_f(float x) {
  return x / (1.0f + __expf(-x));
}

// gfx1250 async copy: global -> LDS, 16 bytes per lane, tracked by ASYNCcnt.
__device__ __forceinline__ void async_b128(uint32_t lds_off, uint64_t gaddr) {
  asm volatile("global_load_async_to_lds_b128 %0, %1, off"
               :: "v"(lds_off), "v"(gaddr) : "memory");
}
__device__ __forceinline__ void wait_async0() {
  asm volatile("s_wait_asynccnt 0x0" ::: "memory");
}

// A operand: 16xK bf16 row-major matrix (global or LDS), K-chunk base k0.
// ISA 16-bit A layout: lanes 0-15 hold row (lane), K = k0+{0..7,16..23};
// lanes 16-31 hold row (lane-16), K = k0+{8..15,24..31}.
__device__ __forceinline__ bf16x16 load_a_bf16(const bf16* mat, int ld, int k0, int lane) {
  int row = lane & 15;
  int sel = (lane & 16) ? 8 : 0;
  const bf16* p = mat + row * ld + k0 + sel;
  BF16x16 r;
  r.h[0] = *reinterpret_cast<const bf16x8*>(p);
  r.h[1] = *reinterpret_cast<const bf16x8*>(p + 16);
  return r.v;
}

__device__ __forceinline__ bf16x16 load_a_row(const bf16* rowbase, int k0, int lane) {
  int sel = (lane & 16) ? 8 : 0;
  const bf16* p = rowbase + k0 + sel;
  BF16x16 r;
  r.h[0] = *reinterpret_cast<const bf16x8*>(p);
  r.h[1] = *reinterpret_cast<const bf16x8*>(p + 16);
  return r.v;
}

__device__ __forceinline__ bf16x16 load_a_f32row(const float* rowbase, int k0, int lane) {
  int sel = (lane & 16) ? 8 : 0;
  const float* p = rowbase + k0 + sel;
  BF16x16 r;
#pragma unroll
  for (int j = 0; j < 8; ++j) {
    r.v[j]     = (bf16)p[j];
    r.v[j + 8] = (bf16)p[16 + j];
  }
  return r.v;
}

// ---------------- weight repack: f32 [K,128] -> WMMA-B bf16 lane layout ----
__global__ void pack_w(const float* __restrict__ src, bf16* __restrict__ dst, int K) {
  int p = blockIdx.x * blockDim.x + threadIdx.x;
  int total = K * 128;
  if (p >= total) return;
  int j    = p & 15;
  int lane = (p >> 4) & 31;
  int rest = p >> 9;
  int nkt  = K >> 5;
  int kt   = rest % nkt;
  int c    = rest / nkt;
  int kloc = (j < 8 ? j : j + 8) + ((lane & 16) ? 8 : 0);
  int k = kt * 32 + kloc;
  int n = c * 16 + (lane & 15);
  dst[p] = (bf16)src[k * 128 + n];
}

__global__ void init_kernel(const float* __restrict__ pos, float* __restrict__ posA,
                            float* __restrict__ deg) {
  int i = blockIdx.x * blockDim.x + threadIdx.x;
  if (i < NN) deg[i] = 0.f;
  if (i < NN * 3) posA[i] = pos[i];
}

__global__ void deg_kernel(const int* __restrict__ dst, float* __restrict__ deg) {
  int e = blockIdx.x * blockDim.x + threadIdx.x;
  if (e < N_EDGES) atomicAdd(&deg[dst[e]], 1.0f);
}

__global__ void zero_kernel(float* __restrict__ agg, float* __restrict__ posacc) {
  int i = blockIdx.x * blockDim.x + threadIdx.x;
  if (i < NN * HID) agg[i] = 0.f;
  if (i < NN * 3) posacc[i] = 0.f;
}

// ---------------- embedding: h = x @ emb_W + emb_b -----------------------
__global__ void __launch_bounds__(128)
embed_kernel(const float* __restrict__ x, const bf16x16* __restrict__ Wp,
             const float* __restrict__ b, float* __restrict__ h, bf16* __restrict__ hbf) {
  int lane = threadIdx.x & 31;
  int wv   = threadIdx.x >> 5;
  int tile = blockIdx.x * 4 + wv;
  int n0   = tile * 16;
  int mrow = lane & 15;
  int hi8  = (lane & 16) ? 8 : 0;
  const float* rowbase = x + (size_t)(n0 + mrow) * IN_CH;
#pragma unroll
  for (int c = 0; c < 8; ++c) {
    f32x8 acc = f32x8_zero();
#pragma unroll
    for (int kt = 0; kt < 2; ++kt) {
      bf16x16 a = load_a_f32row(rowbase, kt * 32, lane);
      acc = wmma_bf16(a, Wp[(c * 2 + kt) * 32 + lane], acc);
    }
    int col = c * 16 + mrow;
    float bias = b[col];
#pragma unroll
    for (int r = 0; r < 8; ++r) {
      int m = r + hi8;
      float v = acc[r] + bias;
      size_t o = (size_t)(n0 + m) * HID + col;
      h[o] = v;
      hbf[o] = (bf16)v;
    }
  }
}

// ---------------- fused per-edge pipeline (32 edges / wave) --------------
// Stage gathered h rows into LDS via async copies, then two 16-row M-tiles
// share every B operand (half the weight traffic per WMMA).
__global__ void __launch_bounds__(64)
edge_kernel(const int* __restrict__ src, const int* __restrict__ dst,
            const bf16* __restrict__ hbf, const float* __restrict__ posA,
            const bf16x16* __restrict__ W1p, const float* __restrict__ b1,
            const float* __restrict__ w1last,
            const bf16x16* __restrict__ W2p, const float* __restrict__ b2,
            const bf16x16* __restrict__ Pw1p, const float* __restrict__ pb1,
            const float* __restrict__ pw2, const float* __restrict__ pb2,
            float* __restrict__ agg, float* __restrict__ posacc) {
  __shared__ int   s_di[2][32];
  __shared__ float s_rel[2][32][3], s_d2[2][32], s_w[2][32];
  __shared__ __align__(16) bf16 s_hd[2][32 * HID];  // dst rows; reused for m
  __shared__ __align__(16) bf16 s_hs[2][32 * HID];  // src rows
  __shared__ __align__(16) bf16 s_t1[2][32 * HID];  // hidden of edge MLP

  int lane = threadIdx.x & 31;
  int wv   = threadIdx.x >> 5;
  int e0   = (blockIdx.x * 2 + wv) * 32;

  {
    int e  = e0 + lane;
    int si = src[e], di = dst[e];
    s_di[wv][lane] = di;
    float rx = posA[di * 3 + 0] - posA[si * 3 + 0];
    float ry = posA[di * 3 + 1] - posA[si * 3 + 1];
    float rz = posA[di * 3 + 2] - posA[si * 3 + 2];
    s_rel[wv][lane][0] = rx;
    s_rel[wv][lane][1] = ry;
    s_rel[wv][lane][2] = rz;
    s_d2[wv][lane] = rx * rx + ry * ry + rz * rz;
    // async-stage this lane's gathered rows (256 B each) into LDS
    uint64_t gd = (uint64_t)(uintptr_t)(hbf + (size_t)di * HID);
    uint64_t gs = (uint64_t)(uintptr_t)(hbf + (size_t)si * HID);
    uint32_t ld = (uint32_t)(uintptr_t)&s_hd[wv][lane * HID];
    uint32_t ls = (uint32_t)(uintptr_t)&s_hs[wv][lane * HID];
#pragma unroll
    for (int j = 0; j < 16; ++j) {
      async_b128(ld + j * 16, gd + j * 16);
      async_b128(ls + j * 16, gs + j * 16);
    }
  }
  wait_async0();
  __syncthreads();

  int mrow = lane & 15;
  int hi8  = (lane & 16) ? 8 : 0;

  // ---- GEMM1: [32,256] @ [256,128] (+ d2 rank-1 term), SiLU -> s_t1
#pragma unroll
  for (int c = 0; c < 8; ++c) {
    f32x8 acc0 = f32x8_zero();
    f32x8 acc1 = f32x8_zero();
#pragma unroll
    for (int kt = 0; kt < 4; ++kt) {
      bf16x16 b = W1p[(c * 8 + kt) * 32 + lane];
      acc0 = wmma_bf16(load_a_bf16(&s_hd[wv][0], HID, kt * 32, lane), b, acc0);
      acc1 = wmma_bf16(load_a_bf16(&s_hd[wv][16 * HID], HID, kt * 32, lane), b, acc1);
    }
#pragma unroll
    for (int kt = 0; kt < 4; ++kt) {
      bf16x16 b = W1p[(c * 8 + 4 + kt) * 32 + lane];
      acc0 = wmma_bf16(load_a_bf16(&s_hs[wv][0], HID, kt * 32, lane), b, acc0);
      acc1 = wmma_bf16(load_a_bf16(&s_hs[wv][16 * HID], HID, kt * 32, lane), b, acc1);
    }
    int col = c * 16 + mrow;
    float bias = b1[col];
    float wl   = w1last[col];
#pragma unroll
    for (int r = 0; r < 8; ++r) {
      int m0 = r + hi8;
      int m1 = 16 + r + hi8;
      s_t1[wv][m0 * HID + col] = (bf16)silu_f(acc0[r] + bias + s_d2[wv][m0] * wl);
      s_t1[wv][m1 * HID + col] = (bf16)silu_f(acc1[r] + bias + s_d2[wv][m1] * wl);
    }
  }
  __syncthreads();

  // ---- GEMM2: [32,128] @ [128,128], SiLU -> m (into s_hd) ; scatter agg
#pragma unroll
  for (int c = 0; c < 8; ++c) {
    f32x8 acc0 = f32x8_zero();
    f32x8 acc1 = f32x8_zero();
#pragma unroll
    for (int kt = 0; kt < 4; ++kt) {
      bf16x16 b = W2p[(c * 4 + kt) * 32 + lane];
      acc0 = wmma_bf16(load_a_bf16(&s_t1[wv][0], HID, kt * 32, lane), b, acc0);
      acc1 = wmma_bf16(load_a_bf16(&s_t1[wv][16 * HID], HID, kt * 32, lane), b, acc1);
    }
    int col = c * 16 + mrow;
    float bias = b2[col];
#pragma unroll
    for (int r = 0; r < 8; ++r) {
      int m0 = r + hi8;
      int m1 = 16 + r + hi8;
      float v0 = silu_f(acc0[r] + bias);
      float v1 = silu_f(acc1[r] + bias);
      s_hd[wv][m0 * HID + col] = (bf16)v0;
      s_hd[wv][m1 * HID + col] = (bf16)v1;
      atomicAdd(&agg[(size_t)s_di[wv][m0] * HID + col], v0);
      atomicAdd(&agg[(size_t)s_di[wv][m1] * HID + col], v1);
    }
  }
  __syncthreads();

  // ---- pos MLP: t = silu(m @ pW1 + pb1); w = t @ pW2 + pb2 (per-edge scalar)
  float wp0[8] = {0.f, 0.f, 0.f, 0.f, 0.f, 0.f, 0.f, 0.f};
  float wp1[8] = {0.f, 0.f, 0.f, 0.f, 0.f, 0.f, 0.f, 0.f};
#pragma unroll
  for (int c = 0; c < 8; ++c) {
    f32x8 acc0 = f32x8_zero();
    f32x8 acc1 = f32x8_zero();
#pragma unroll
    for (int kt = 0; kt < 4; ++kt) {
      bf16x16 b = Pw1p[(c * 4 + kt) * 32 + lane];
      acc0 = wmma_bf16(load_a_bf16(&s_hd[wv][0], HID, kt * 32, lane), b, acc0);
      acc1 = wmma_bf16(load_a_bf16(&s_hd[wv][16 * HID], HID, kt * 32, lane), b, acc1);
    }
    int col = c * 16 + mrow;
    float bias = pb1[col];
    float w2c  = pw2[col];
#pragma unroll
    for (int r = 0; r < 8; ++r) {
      wp0[r] += silu_f(acc0[r] + bias) * w2c;
      wp1[r] += silu_f(acc1[r] + bias) * w2c;
    }
  }
  // reduce over the 16 N-lanes of each half-wave (wave32)
#pragma unroll
  for (int r = 0; r < 8; ++r) {
#pragma unroll
    for (int off = 8; off >= 1; off >>= 1) {
      wp0[r] += __shfl_xor(wp0[r], off, 16);
      wp1[r] += __shfl_xor(wp1[r], off, 16);
    }
  }
  if (mrow == 0) {
    float pb = pb2[0];
#pragma unroll
    for (int r = 0; r < 8; ++r) {
      s_w[wv][r + hi8]      = wp0[r] + pb;
      s_w[wv][16 + r + hi8] = wp1[r] + pb;
    }
  }
  __syncthreads();

  {
    int di  = s_di[wv][lane];
    float w = s_w[wv][lane];
    atomicAdd(&posacc[di * 3 + 0], s_rel[wv][lane][0] * w);
    atomicAdd(&posacc[di * 3 + 1], s_rel[wv][lane][1] * w);
    atomicAdd(&posacc[di * 3 + 2], s_rel[wv][lane][2] * w);
  }
}

// ---------------- node update -------------------------------------------
__global__ void __launch_bounds__(128)
node_kernel(const float* __restrict__ agg,
            const bf16x16* __restrict__ W1p, const float* __restrict__ b1,
            const bf16x16* __restrict__ W2p, const float* __restrict__ b2,
            float* __restrict__ h, bf16* __restrict__ hbf,
            const float* __restrict__ posacc, const float* __restrict__ deg,
            float* __restrict__ posA) {
  __shared__ __align__(16) bf16 s_t1[4][16 * HID];
  int lane = threadIdx.x & 31;
  int wv   = threadIdx.x >> 5;
  int tile = blockIdx.x * 4 + wv;
  int n0   = tile * 16;
  int mrow = lane & 15;
  int hi8  = (lane & 16) ? 8 : 0;
  const bf16*  rowH = hbf + (size_t)(n0 + mrow) * HID;
  const float* rowA = agg + (size_t)(n0 + mrow) * HID;

#pragma unroll
  for (int c = 0; c < 8; ++c) {
    f32x8 acc = f32x8_zero();
#pragma unroll
    for (int kt = 0; kt < 4; ++kt)
      acc = wmma_bf16(load_a_row(rowH, kt * 32, lane), W1p[(c * 8 + kt) * 32 + lane], acc);
#pragma unroll
    for (int kt = 0; kt < 4; ++kt)
      acc = wmma_bf16(load_a_f32row(rowA, kt * 32, lane), W1p[(c * 8 + 4 + kt) * 32 + lane], acc);
    int col = c * 16 + mrow;
    float bias = b1[col];
#pragma unroll
    for (int r = 0; r < 8; ++r) {
      int m = r + hi8;
      s_t1[wv][m * HID + col] = (bf16)silu_f(acc[r] + bias);
    }
  }
  __syncthreads();

#pragma unroll
  for (int c = 0; c < 8; ++c) {
    f32x8 acc = f32x8_zero();
#pragma unroll
    for (int kt = 0; kt < 4; ++kt)
      acc = wmma_bf16(load_a_bf16(&s_t1[wv][0], HID, kt * 32, lane),
                      W2p[(c * 4 + kt) * 32 + lane], acc);
    int col = c * 16 + mrow;
    float bias = b2[col];
#pragma unroll
    for (int r = 0; r < 8; ++r) {
      int m = r + hi8;
      float v = acc[r] + bias;  // linear output
      size_t o = (size_t)(n0 + m) * HID + col;
      h[o] = v;
      hbf[o] = (bf16)v;
    }
  }

  if (lane < 16) {
    int n = n0 + lane;
    float dinv = 1.0f / fmaxf(deg[n], 1.0f);
#pragma unroll
    for (int k = 0; k < 3; ++k)
      posA[n * 3 + k] += posacc[n * 3 + k] * dinv;
  }
}

__global__ void finalize_kernel(const float* __restrict__ h, const float* __restrict__ posA,
                                float* __restrict__ out) {
  int i = blockIdx.x * blockDim.x + threadIdx.x;
  if (i < NN * HID) out[i] = h[i];
  if (i < NN * 3) out[NN * HID + i] = posA[i];
}

extern "C" void kernel_launch(void* const* d_in, const int* in_sizes, int n_in,
                              void* d_out, int out_size, void* d_ws, size_t ws_size,
                              hipStream_t stream) {
  (void)in_sizes; (void)n_in; (void)out_size; (void)ws_size;
  const int*   eidx  = (const int*)d_in[0];
  const int*   src   = eidx;
  const int*   dstp  = eidx + N_EDGES;
  const float* x     = (const float*)d_in[1];
  const float* pos   = (const float*)d_in[2];
  const float* emb_W = (const float*)d_in[3];
  const float* emb_b = (const float*)d_in[4];
  const float* eW1   = (const float*)d_in[5];
  const float* eb1   = (const float*)d_in[6];
  const float* eW2   = (const float*)d_in[7];
  const float* eb2   = (const float*)d_in[8];
  const float* nW1   = (const float*)d_in[9];
  const float* nb1   = (const float*)d_in[10];
  const float* nW2   = (const float*)d_in[11];
  const float* nb2   = (const float*)d_in[12];
  const float* pW1   = (const float*)d_in[13];
  const float* pb1   = (const float*)d_in[14];
  const float* pW2   = (const float*)d_in[15];
  const float* pb2   = (const float*)d_in[16];

  char* ws = (char*)d_ws;
  bf16*  hbf    = (bf16*)ws;  ws += (size_t)NN * HID * 2;
  float* h      = (float*)ws; ws += (size_t)NN * HID * 4;
  float* agg    = (float*)ws; ws += (size_t)NN * HID * 4;
  float* posA   = (float*)ws; ws += (size_t)NN * 3 * 4;
  float* posacc = (float*)ws; ws += (size_t)NN * 3 * 4;
  float* deg    = (float*)ws; ws += (size_t)NN * 4;
  bf16* embP  = (bf16*)ws; ws += (size_t)64 * 128 * 2;
  bf16* eW1P  = (bf16*)ws; ws += (size_t)3 * 256 * 128 * 2;
  bf16* eW2P  = (bf16*)ws; ws += (size_t)3 * 128 * 128 * 2;
  bf16* nW1P  = (bf16*)ws; ws += (size_t)3 * 256 * 128 * 2;
  bf16* nW2P  = (bf16*)ws; ws += (size_t)3 * 128 * 128 * 2;
  bf16* pW1P  = (bf16*)ws; ws += (size_t)3 * 128 * 128 * 2;

  pack_w<<<(64 * 128 + 255) / 256, 256, 0, stream>>>(emb_W, embP, 64);
  for (int l = 0; l < 3; ++l) {
    pack_w<<<(256 * 128 + 255) / 256, 256, 0, stream>>>(eW1 + (size_t)l * 257 * 128,
                                                        eW1P + (size_t)l * 256 * 128, 256);
    pack_w<<<(128 * 128 + 255) / 256, 256, 0, stream>>>(eW2 + (size_t)l * 128 * 128,
                                                        eW2P + (size_t)l * 128 * 128, 128);
    pack_w<<<(256 * 128 + 255) / 256, 256, 0, stream>>>(nW1 + (size_t)l * 256 * 128,
                                                        nW1P + (size_t)l * 256 * 128, 256);
    pack_w<<<(128 * 128 + 255) / 256, 256, 0, stream>>>(nW2 + (size_t)l * 128 * 128,
                                                        nW2P + (size_t)l * 128 * 128, 128);
    pack_w<<<(128 * 128 + 255) / 256, 256, 0, stream>>>(pW1 + (size_t)l * 128 * 128,
                                                        pW1P + (size_t)l * 128 * 128, 128);
  }

  init_kernel<<<(NN * 3 + 255) / 256, 256, 0, stream>>>(pos, posA, deg);
  deg_kernel<<<(N_EDGES + 255) / 256, 256, 0, stream>>>(dstp, deg);
  embed_kernel<<<NN / 64, 128, 0, stream>>>(x, (const bf16x16*)embP, emb_b, h, hbf);

  for (int l = 0; l < 3; ++l) {
    zero_kernel<<<(NN * HID + 255) / 256, 256, 0, stream>>>(agg, posacc);
    edge_kernel<<<N_EDGES / 64, 64, 0, stream>>>(
        src, dstp, hbf, posA,
        (const bf16x16*)(eW1P + (size_t)l * 256 * 128), eb1 + l * 128,
        eW1 + (size_t)l * 257 * 128 + 256 * 128,
        (const bf16x16*)(eW2P + (size_t)l * 128 * 128), eb2 + l * 128,
        (const bf16x16*)(pW1P + (size_t)l * 128 * 128), pb1 + l * 128,
        pW2 + (size_t)l * 128, pb2 + l,
        agg, posacc);
    node_kernel<<<NN / 64, 128, 0, stream>>>(
        agg,
        (const bf16x16*)(nW1P + (size_t)l * 256 * 128), nb1 + l * 128,
        (const bf16x16*)(nW2P + (size_t)l * 128 * 128), nb2 + l * 128,
        h, hbf, posacc, deg, posA);
  }

  finalize_kernel<<<(NN * HID + 255) / 256, 256, 0, stream>>>(h, posA, (float*)d_out);
}
